// JetGNN_28295244546252
// MI455X (gfx1250) — compile-verified, matrix operand order checked by reading, not run on hardware
//
#include <hip/hip_runtime.h>
#include <hip/hip_bf16.h>

typedef __bf16 bf16;
typedef __attribute__((ext_vector_type(16))) __bf16 v16bf;
typedef __attribute__((ext_vector_type(8)))  float  v8f;

#define NNODES 32768
#define NEDGES (NNODES * 16)
#define NGRAPH 64
#define EPSBN  1e-5f

static inline int ru32(int x) { return (x + 31) & ~31; }

// ---------------------------------------------------------------------------
// order-preserving float <-> uint encoding for atomicMax-based segment max
// sentinel 0u == "never written" -> decodes to 0.0f (matches isfinite fixup)
// ---------------------------------------------------------------------------
__device__ inline unsigned enc_f(float f) {
  int i = __float_as_int(f);
  return (i >= 0) ? ((unsigned)i | 0x80000000u) : ~(unsigned)i;
}
__device__ inline float dec_f(unsigned u) {
  if (u == 0u) return 0.0f;
  int i = (u & 0x80000000u) ? (int)(u & 0x7fffffffu) : ~(int)u;
  return __int_as_float(i);
}
__device__ inline float lrelu(float v) { return v > 0.f ? v : 0.2f * v; }

// ---------------------------------------------------------------------------
// WMMA bf16 GEMM with fused BN-stats (and optional fused affine+lrelu on A):
//   Y[M,N] = (AFF ? lrelu(A*scale[k]+shift[k]) : A) [M,K]  x  Bsw[K,N]
// A row-major bf16 (K%32==0). Bsw pre-swizzled fragment order [K/32][N][32].
// Y written as bf16 (pre-BN). Per-column sum/sumsq accumulated via LDS
// ds_add_f32 then one global atomicAdd per column per block.
// Block: 256 threads = 8 waves; wave = 16-row strip x all N. M%128==0.
// ---------------------------------------------------------------------------
template <int NT, bool AFF>
__launch_bounds__(256, 1)
__global__ void gemm_bf16_wmma(const bf16* __restrict__ A,
                               const bf16* __restrict__ Bsw,
                               bf16* __restrict__ Y,
                               float* __restrict__ gsum, float* __restrict__ gsq,
                               const float* __restrict__ ascale,
                               const float* __restrict__ ashift,
                               int M, int K) {
  constexpr int Ncol = NT * 16;
  __shared__ float cred[2 * Ncol];
  for (int i = threadIdx.x; i < 2 * Ncol; i += 256) cred[i] = 0.f;
  __syncthreads();

  const int lane = threadIdx.x & 31;
  const int wave = threadIdx.x >> 5;
  const int g    = lane >> 4;   // lane half-group: selects K sub-pattern
  const int mr   = lane & 15;   // row within 16-tile (A) / col (B, D)
  const int row0 = (blockIdx.x * 8 + wave) * 16;

  v8f zero = {0.f, 0.f, 0.f, 0.f, 0.f, 0.f, 0.f, 0.f};
  v8f acc[NT];
#pragma unroll
  for (int t = 0; t < NT; ++t) acc[t] = zero;

  union Frag { v16bf v; uint4 q[2]; bf16 h[16]; };

  for (int k0 = 0; k0 < K; k0 += 32) {
    // A fragment: row = row0+mr; element j -> k = k0 + g*8 + (j<8 ? j : 8+j)
    const bf16* arow = A + (size_t)(row0 + mr) * K + k0 + g * 8;
    __builtin_prefetch(arow + 64, 0, 1);
    Frag a;
    a.q[0] = *(const uint4*)(arow);
    a.q[1] = *(const uint4*)(arow + 16);
    if (AFF) {
#pragma unroll
      for (int j = 0; j < 16; ++j) {
        int k = k0 + g * 8 + (j < 8 ? j : 8 + j);
        float v = (float)a.h[j] * ascale[k] + ashift[k];
        a.h[j] = (bf16)lrelu(v);
      }
    }

    const bf16* bbase = Bsw + ((size_t)(k0 >> 5) * Ncol + mr) * 32 + g * 8;
#pragma unroll
    for (int t = 0; t < NT; ++t) {
      const bf16* bp = bbase + (size_t)t * 16 * 32;
      Frag b;
      b.q[0] = *(const uint4*)(bp);
      b.q[1] = *(const uint4*)(bp + 16);
      acc[t] = __builtin_amdgcn_wmma_f32_16x16x32_bf16(
          false, a.v, false, b.v, (short)0, acc[t], false, false);
    }
  }

  // store bf16 output + per-column stats into LDS
#pragma unroll
  for (int t = 0; t < NT; ++t) {
    int c = t * 16 + mr;           // D: N = lane
    float s = 0.f, ss = 0.f;
#pragma unroll
    for (int i = 0; i < 8; ++i) {
      float v = acc[t][i];
      s += v; ss += v * v;
      int r = row0 + i + 8 * g;    // D: VGPR i -> M = i (+8 for hi lanes)
      Y[(size_t)r * Ncol + c] = (bf16)v;
    }
    atomicAdd(&cred[c], s);
    atomicAdd(&cred[Ncol + c], ss);
  }
  __syncthreads();
  for (int i = threadIdx.x; i < Ncol; i += 256) {
    atomicAdd(&gsum[i], cred[i]);
    atomicAdd(&gsq[i], cred[Ncol + i]);
  }
}

// ---------------------------------------------------------------------------
// elementwise / prep kernels
// ---------------------------------------------------------------------------
// weights -> bf16, zero K-pad, swizzled to fragment order [Kpad/32][Nout][32]
__global__ void pad_w_swz(const float* __restrict__ W, bf16* __restrict__ O,
                          int Kin, int Kpad, int Nout) {
  int i = blockIdx.x * blockDim.x + threadIdx.x;
  if (i >= Kpad * Nout) return;
  int kp = i & 31;
  int n  = (i >> 5) % Nout;
  int kb = i / (32 * Nout);
  int k  = kb * 32 + kp;
  O[i] = (k < Kin) ? (bf16)W[k * Nout + n] : (bf16)0.f;
}

__global__ void pad_nodes_bf16(const float* __restrict__ X, bf16* __restrict__ O,
                               int Nn, int C, int Kpad) {
  long long i = (long long)blockIdx.x * blockDim.x + threadIdx.x;
  if (i >= (long long)Nn * Kpad) return;
  int k = (int)(i % Kpad);
  long long n = i / Kpad;
  O[i] = (k < C) ? (bf16)X[n * C + k] : (bf16)0.f;
}

// per-edge feature: [ x[dst], x[src]-x[dst], 0-pad ] -> bf16 [E, Kpad]
// node features are L2-resident (<=32 MB) so the random gathers stay on-chip
__global__ void gather_edges(const float* __restrict__ X, const int* __restrict__ src,
                             const int* __restrict__ dst, bf16* __restrict__ A,
                             int E, int C, int Kpad) {
  int e = blockIdx.x * blockDim.x + threadIdx.x;
  if (e >= E) return;
  int s = src[e], d = dst[e];
  const float* xs = X + (size_t)s * C;
  const float* xd = X + (size_t)d * C;
  bf16* row = A + (size_t)e * Kpad;
  for (int c = 0; c < C; ++c) {
    float xi = xd[c], xj = xs[c];
    row[c]     = (bf16)xi;
    row[C + c] = (bf16)(xj - xi);
  }
  for (int c = 2 * C; c < Kpad; ++c) row[c] = (bf16)0.f;
}

// fold BN into per-channel scale/shift: scale=g*rsqrt(var+eps), shift=b-mean*scale
__global__ void bn_finalize2(const float* __restrict__ sum, const float* __restrict__ sumsq,
                             const float* __restrict__ gam, const float* __restrict__ bet,
                             float* __restrict__ scale, float* __restrict__ shift,
                             float Mf, int Nc) {
  int c = blockIdx.x * blockDim.x + threadIdx.x;
  if (c >= Nc) return;
  float m  = sum[c] / Mf;
  float v  = sumsq[c] / Mf - m * m;
  float sc = rsqrtf(v + EPSBN) * gam[c];
  scale[c] = sc;
  shift[c] = bet[c] - m * sc;
}

__global__ void norm_scatter_max(const bf16* __restrict__ Y, const float* __restrict__ scale,
                                 const float* __restrict__ shift, const int* __restrict__ dst,
                                 unsigned* __restrict__ agg, long long total, int Nc) {
  long long i = (long long)blockIdx.x * blockDim.x + threadIdx.x;
  if (i >= total) return;
  int c = (int)(i % Nc);
  long long r = i / Nc;
  float v = lrelu((float)Y[i] * scale[c] + shift[c]);
  atomicMax(&agg[(size_t)dst[r] * Nc + c], enc_f(v));
}

__global__ void combine_skip(const unsigned* __restrict__ agg, const bf16* __restrict__ Ys,
                             const float* __restrict__ scale, const float* __restrict__ shift,
                             float* __restrict__ xout, long long total, int Nc) {
  long long i = (long long)blockIdx.x * blockDim.x + threadIdx.x;
  if (i >= total) return;
  int c = (int)(i % Nc);
  float a = dec_f(agg[i]);
  float s = (float)Ys[i] * scale[c] + shift[c];
  xout[i] = lrelu(a + s);
}

// ---------------------------------------------------------------------------
// per-graph pooling
// ---------------------------------------------------------------------------
__global__ void pool_count(const int* __restrict__ batch, float* __restrict__ counts) {
  int n = blockIdx.x * blockDim.x + threadIdx.x;
  if (n >= NNODES) return;
  atomicAdd(&counts[batch[n]], 1.0f);
}

__global__ void pool_accum(const float* __restrict__ X, const int* __restrict__ batch,
                           float* __restrict__ meanacc, unsigned* __restrict__ maxenc) {
  long long i = (long long)blockIdx.x * blockDim.x + threadIdx.x;
  if (i >= (long long)NNODES * 256) return;
  int n = (int)(i >> 8), c = (int)(i & 255);
  int gph = batch[n];
  float v = X[i];
  atomicAdd(&meanacc[gph * 256 + c], v);
  atomicMax(&maxenc[gph * 256 + c], enc_f(v));
}

__global__ void pool_final(const float* __restrict__ meanacc, const float* __restrict__ counts,
                           const unsigned* __restrict__ maxenc, float* __restrict__ Z) {
  int i = blockIdx.x * blockDim.x + threadIdx.x;
  if (i >= NGRAPH * 256) return;
  int gph = i >> 8, c = i & 255;
  float cnt = fmaxf(counts[gph], 1.0f);
  Z[gph * 512 + c]       = meanacc[i] / cnt;
  Z[gph * 512 + 256 + c] = dec_f(maxenc[i]);
}

// ---------------------------------------------------------------------------
// tiny classifier head (G=64 rows)
// ---------------------------------------------------------------------------
__global__ void fc_kernel(const float* __restrict__ Z, const float* __restrict__ W,
                          const float* __restrict__ bias, float* __restrict__ out,
                          int G, int Kin, int Nout) {
  int i = blockIdx.x * blockDim.x + threadIdx.x;
  if (i >= G * Nout) return;
  int g = i / Nout, j = i % Nout;
  float s = bias[j];
  for (int k = 0; k < Kin; ++k) s += Z[g * Kin + k] * W[k * Nout + j];
  out[i] = s;
}

__global__ void bn_lrelu_cols(float* __restrict__ Y, const float* __restrict__ gam,
                              const float* __restrict__ bet, int G, int Nc) {
  int j = blockIdx.x * blockDim.x + threadIdx.x;
  if (j >= Nc) return;
  float s = 0.f, ss = 0.f;
  for (int g = 0; g < G; ++g) { float v = Y[g * Nc + j]; s += v; ss += v * v; }
  float m = s / G;
  float istd = rsqrtf(ss / G - m * m + EPSBN);
  for (int g = 0; g < G; ++g) {
    float v = (Y[g * Nc + j] - m) * istd * gam[j] + bet[j];
    Y[g * Nc + j] = lrelu(v);
  }
}

// ---------------------------------------------------------------------------
// host orchestration
// ---------------------------------------------------------------------------
extern "C" void kernel_launch(void* const* d_in, const int* in_sizes, int n_in,
                              void* d_out, int out_size, void* d_ws, size_t ws_size,
                              hipStream_t stream) {
  (void)in_sizes; (void)n_in; (void)out_size; (void)ws_size;
  const float* x_in  = (const float*)d_in[0];
  const int*   ei    = (const int*)d_in[1];
  const int*   src   = ei;
  const int*   dst   = ei + NEDGES;
  const int*   batch = (const int*)d_in[2];

  struct Blk { const float *w1,*g1,*b1,*w2,*g2,*b2,*ws,*gs,*bs; };
  Blk bl[3];
  int p = 3;
  for (int i = 0; i < 3; ++i) {
    bl[i].w1 = (const float*)d_in[p++]; bl[i].g1 = (const float*)d_in[p++];
    bl[i].b1 = (const float*)d_in[p++]; bl[i].w2 = (const float*)d_in[p++];
    bl[i].g2 = (const float*)d_in[p++]; bl[i].b2 = (const float*)d_in[p++];
    bl[i].ws = (const float*)d_in[p++]; bl[i].gs = (const float*)d_in[p++];
    bl[i].bs = (const float*)d_in[p++];
  }
  const float* fc1_w = (const float*)d_in[p++]; const float* fc1_b = (const float*)d_in[p++];
  const float* bn1_g = (const float*)d_in[p++]; const float* bn1_b = (const float*)d_in[p++];
  const float* fc2_w = (const float*)d_in[p++]; const float* fc2_b = (const float*)d_in[p++];
  const float* bn2_g = (const float*)d_in[p++]; const float* bn2_b = (const float*)d_in[p++];
  const float* fc3_w = (const float*)d_in[p++]; const float* fc3_b = (const float*)d_in[p++];

  // ---- workspace layout ----
  char* W = (char*)d_ws;
  size_t off = 0;
  auto take = [&](size_t b) -> void* {
    void* ptr = W + off;
    off += (b + 255) & ~(size_t)255;
    return ptr;
  };
  bf16*     AE   = (bf16*)take((size_t)NEDGES * 256 * 2);
  bf16*     Y1   = (bf16*)take((size_t)NEDGES * 256 * 2);
  bf16*     Y2   = (bf16*)take((size_t)NEDGES * 256 * 2);
  float*    XA   = (float*)take((size_t)NNODES * 256 * 4);
  float*    XB   = (float*)take((size_t)NNODES * 256 * 4);
  bf16*     XBF  = (bf16*)take((size_t)NNODES * 256 * 2);
  unsigned* AGG  = (unsigned*)take((size_t)NNODES * 256 * 4);
  bf16*     YS   = (bf16*)take((size_t)NNODES * 256 * 2);
  float*    STAT = (float*)take(4 * 256 * 4);
  float *gsum = STAT, *gsq = STAT + 256, *scale = STAT + 512, *shift = STAT + 768;
  bf16*     WB   = (bf16*)take(1u << 20);
  float*    MACC = (float*)take(NGRAPH * 256 * 4);
  float*    CNT  = (float*)take(NGRAPH * 4);
  unsigned* MENC = (unsigned*)take(NGRAPH * 256 * 4);
  float*    Z    = (float*)take(NGRAPH * 512 * 4);
  float*    Z1   = (float*)take(NGRAPH * 256 * 4);
  float*    Z2   = (float*)take(NGRAPH * 128 * 4);

  const int cin[3]  = {7, 64, 128};
  const int cout[3] = {64, 128, 256};

  // bf16 swizzled weight slots
  bf16 *w1b[3], *w2b[3], *wsb[3];
  size_t wo = 0;
  for (int i = 0; i < 3; ++i) {
    int Kp1 = ru32(2 * cin[i]), Kps = ru32(cin[i]);
    w1b[i] = WB + wo; wo += (size_t)Kp1 * cout[i];
    w2b[i] = WB + wo; wo += (size_t)cout[i] * cout[i];
    wsb[i] = WB + wo; wo += (size_t)Kps * cout[i];
  }

  auto launch_gemm = [&](int NT, bool aff, const bf16* A, const bf16* Bm, bf16* Y,
                         const float* asc, const float* ash, int M, int K) {
    hipMemsetAsync(gsum, 0, 2 * 256 * 4, stream);   // zero sum+sumsq
    dim3 grid(M / 128), blk(256);
    if (NT == 4) {
      if (aff) gemm_bf16_wmma<4,  true ><<<grid, blk, 0, stream>>>(A, Bm, Y, gsum, gsq, asc, ash, M, K);
      else     gemm_bf16_wmma<4,  false><<<grid, blk, 0, stream>>>(A, Bm, Y, gsum, gsq, asc, ash, M, K);
    } else if (NT == 8) {
      if (aff) gemm_bf16_wmma<8,  true ><<<grid, blk, 0, stream>>>(A, Bm, Y, gsum, gsq, asc, ash, M, K);
      else     gemm_bf16_wmma<8,  false><<<grid, blk, 0, stream>>>(A, Bm, Y, gsum, gsq, asc, ash, M, K);
    } else {
      if (aff) gemm_bf16_wmma<16, true ><<<grid, blk, 0, stream>>>(A, Bm, Y, gsum, gsq, asc, ash, M, K);
      else     gemm_bf16_wmma<16, false><<<grid, blk, 0, stream>>>(A, Bm, Y, gsum, gsq, asc, ash, M, K);
    }
  };

  // ---- convert/pad/swizzle weights to bf16 fragment order ----
  for (int i = 0; i < 3; ++i) {
    int Kp1 = ru32(2 * cin[i]), Kps = ru32(cin[i]);
    pad_w_swz<<<(Kp1 * cout[i] + 255) / 256, 256, 0, stream>>>(bl[i].w1, w1b[i], 2 * cin[i], Kp1, cout[i]);
    pad_w_swz<<<(cout[i] * cout[i] + 255) / 256, 256, 0, stream>>>(bl[i].w2, w2b[i], cout[i], cout[i], cout[i]);
    pad_w_swz<<<(Kps * cout[i] + 255) / 256, 256, 0, stream>>>(bl[i].ws, wsb[i], cin[i], Kps, cout[i]);
  }

  // ---- EdgeConv blocks ----
  const float* xcur = x_in;
  float* xouts[3] = {XA, XB, XA};
  for (int i = 0; i < 3; ++i) {
    int Cin = cin[i], Cout = cout[i], NT = Cout / 16;
    int Kp1 = ru32(2 * Cin), Kps = ru32(Cin);
    long long totE = (long long)NEDGES * Cout;
    long long totN = (long long)NNODES * Cout;

    gather_edges<<<(NEDGES + 255) / 256, 256, 0, stream>>>(xcur, src, dst, AE, NEDGES, Cin, Kp1);

    // layer 1: GEMM + fused stats
    launch_gemm(NT, false, AE, w1b[i], Y1, nullptr, nullptr, NEDGES, Kp1);
    bn_finalize2<<<1, 256, 0, stream>>>(gsum, gsq, bl[i].g1, bl[i].b1, scale, shift,
                                        (float)NEDGES, Cout);
    // layer 2: BN+lrelu of layer1 fused into A-load; GEMM + fused stats
    launch_gemm(NT, true, Y1, w2b[i], Y2, scale, shift, NEDGES, Cout);
    bn_finalize2<<<1, 256, 0, stream>>>(gsum, gsq, bl[i].g2, bl[i].b2, scale, shift,
                                        (float)NEDGES, Cout);
    hipMemsetAsync(AGG, 0, (size_t)NNODES * Cout * 4, stream);
    norm_scatter_max<<<(unsigned)((totE + 255) / 256), 256, 0, stream>>>(
        Y2, scale, shift, dst, AGG, totE, Cout);

    // skip projection + BN
    pad_nodes_bf16<<<(unsigned)(((long long)NNODES * Kps + 255) / 256), 256, 0, stream>>>(
        xcur, XBF, NNODES, Cin, Kps);
    launch_gemm(NT, false, XBF, wsb[i], YS, nullptr, nullptr, NNODES, Kps);
    bn_finalize2<<<1, 256, 0, stream>>>(gsum, gsq, bl[i].gs, bl[i].bs, scale, shift,
                                        (float)NNODES, Cout);
    combine_skip<<<(unsigned)((totN + 255) / 256), 256, 0, stream>>>(
        AGG, YS, scale, shift, xouts[i], totN, Cout);

    xcur = xouts[i];
  }

  // ---- global mean+max pool per graph ----
  hipMemsetAsync(MACC, 0, NGRAPH * 256 * 4, stream);
  hipMemsetAsync(CNT, 0, NGRAPH * 4, stream);
  hipMemsetAsync(MENC, 0, NGRAPH * 256 * 4, stream);
  pool_count<<<(NNODES + 255) / 256, 256, 0, stream>>>(batch, CNT);
  pool_accum<<<(unsigned)(((long long)NNODES * 256 + 255) / 256), 256, 0, stream>>>(
      xcur, batch, MACC, MENC);
  pool_final<<<(NGRAPH * 256 + 255) / 256, 256, 0, stream>>>(MACC, CNT, MENC, Z);

  // ---- classifier head ----
  fc_kernel<<<(NGRAPH * 256 + 255) / 256, 256, 0, stream>>>(Z, fc1_w, fc1_b, Z1, NGRAPH, 512, 256);
  bn_lrelu_cols<<<1, 256, 0, stream>>>(Z1, bn1_g, bn1_b, NGRAPH, 256);
  fc_kernel<<<(NGRAPH * 128 + 255) / 256, 256, 0, stream>>>(Z1, fc2_w, fc2_b, Z2, NGRAPH, 256, 128);
  bn_lrelu_cols<<<1, 128, 0, stream>>>(Z2, bn2_g, bn2_b, NGRAPH, 128);
  fc_kernel<<<1, 128, 0, stream>>>(Z2, fc3_w, fc3_b, (float*)d_out, NGRAPH, 128, 2);
}
